// Qwen3MoeDecoderLayer_3109556322628
// MI455X (gfx1250) — matrix operand order repaired and checked by reading, not compiled, not run on hardware
//
#include <hip/hip_runtime.h>

typedef __bf16 bf16_t;
typedef __attribute__((ext_vector_type(16))) __bf16 v16bf;
typedef __attribute__((ext_vector_type(8)))  float  v8f;
typedef unsigned int u32;
typedef __attribute__((ext_vector_type(4))) unsigned int v4u;
typedef __attribute__((ext_vector_type(8))) int         v8i;
typedef __attribute__((ext_vector_type(4))) int         v4i;

// ---------------- model dims ----------------
constexpr int BB   = 2;
constexpr int S_   = 1024;
constexpr int D_   = 2048;
constexpr int H_   = 16;
constexpr int KVH_ = 4;
constexpr int HD_  = 128;
constexpr int E_   = 64;
constexpr int KT_  = 8;
constexpr int F_   = 768;
constexpr int T_   = BB * S_;

#define LDA 40     // LDS stride (halves) for 32-wide K tiles: 80B, 16B aligned, conflict-free
#define LQ  136    // LDS stride for 128-wide HD tiles: 272B, 16B aligned, conflict-free

// ---------------- WMMA helpers ----------------
union FragCvt { v16bf v; uint4 q[2]; };

__device__ __forceinline__ v8f zero8() {
  v8f z;
#pragma unroll
  for (int i = 0; i < 8; ++i) z[i] = 0.0f;
  return z;
}

__device__ __forceinline__ v8f wmma_bf16(v16bf a, v16bf b, v8f c) {
  return __builtin_amdgcn_wmma_f32_16x16x32_bf16(false, a, false, b, (short)0, c, false, false);
}

// A fragment (16x32, row-major in LDS): lane<16: row=l, k={0..7,16..23}; lane>=16: row=l-16, k={8..15,24..31}
__device__ __forceinline__ v16bf load_a(const bf16_t* base, int stride, int lane) {
  int r = lane & 15, hi = lane >> 4;
  const bf16_t* p = base + r * stride + 8 * hi;
  FragCvt f;
  f.q[0] = *(const uint4*)(p);
  f.q[1] = *(const uint4*)(p + 16);
  return f.v;
}

// B fragment (32x16) from transposed LDS [n][k]: lane<16: n=l, k={0..7,16..23}; lane>=16: n=l-16, k={8..15,24..31}
__device__ __forceinline__ v16bf load_bt(const bf16_t* base, int stride, int lane) {
  int n = lane & 15, hi = lane >> 4;
  const bf16_t* p = base + n * stride + 8 * hi;
  FragCvt f;
  f.q[0] = *(const uint4*)(p);
  f.q[1] = *(const uint4*)(p + 16);
  return f.v;
}

__device__ __forceinline__ float rowmax16(float v) {
#pragma unroll
  for (int m = 8; m; m >>= 1) v = fmaxf(v, __shfl_xor(v, m, 16));
  return v;
}
__device__ __forceinline__ float rowsum16(float v) {
#pragma unroll
  for (int m = 8; m; m >>= 1) v += __shfl_xor(v, m, 16);
  return v;
}

// ---------------- Tensor Data Mover: 2D bf16 tile global -> LDS ----------------
// D# per CDNA5 ISA. data_size=2B, 2D tile, optional LDS row padding.
// pad_int: pad_interval code (0=2 DW,1=4,2=8,3=16,4=32,5=64); pad_amt: DWORDs-1 of padding.
// Toolchain uses the 6-arg builtin: (v4u g0, v8i g1, v4i g2, v4i g3, v8i, i32 cpol).
__device__ __forceinline__ void tdm_load_bf16_2d(u32 lds_addr, const void* gptr,
                                                 u32 tile0, u32 tile1, u32 tdim1,
                                                 u32 stride0, u32 pad_int, u32 pad_amt) {
  unsigned long long ga = (unsigned long long)(uintptr_t)gptr;
  v4u g0;
  g0[0] = 1u;                                   // count=1 (valid user descriptor)
  g0[1] = lds_addr;                             // LDS byte address
  g0[2] = (u32)ga;                              // global_addr[31:0]
  g0[3] = (u32)(ga >> 32) | (2u << 30);         // global_addr[56:32] | type=2 ("image")
  v8i g1;
  g1[0] = (int)((1u << 16)                      // data_size = 2 bytes
              | (1u << 20)                      // pad_enable
              | (pad_int << 22) | (pad_amt << 25));
  g1[1] = (int)((tile0 & 0xFFFFu) << 16);       // tensor_dim0[15:0] = tile0
  g1[2] = (int)((tile0 >> 16) | ((tdim1 & 0xFFFFu) << 16)); // tensor_dim0 hi | tensor_dim1 lo
  g1[3] = (int)((tdim1 >> 16) | (tile0 << 16)); // tensor_dim1 hi | tile_dim0
  g1[4] = (int)(tile1 & 0xFFFFu);               // tile_dim1 (tile_dim2 = 0)
  g1[5] = (int)stride0;                         // tensor_dim0_stride[31:0] (elements)
  g1[6] = 0;                                    // stride hi | tensor_dim1_stride lo
  g1[7] = 0;
  v4i z4 = {0, 0, 0, 0};
  v8i z8;
#pragma unroll
  for (int i = 0; i < 8; ++i) z8[i] = 0;
  __builtin_amdgcn_tensor_load_to_lds(g0, g1, z4, z4, z8, 0);
}

__device__ __forceinline__ u32 lds_addr_of(const void* p) {
  return (u32)(uintptr_t)p;
}

// ---------------- elementwise / norm kernels ----------------
__global__ __launch_bounds__(256) void k_rmsnorm_bf16(const float* __restrict__ x,
                                                      const float* __restrict__ w,
                                                      bf16_t* __restrict__ out) {
  int t = blockIdx.x, tid = threadIdx.x, lane = tid & 31, wv = tid >> 5;
  const float* row = x + (size_t)t * D_;
  float ssq = 0.f;
  for (int d = tid; d < D_; d += 256) { float v = row[d]; ssq += v * v; }
#pragma unroll
  for (int m = 16; m; m >>= 1) ssq += __shfl_xor(ssq, m, 32);
  __shared__ float red[8];
  if (lane == 0) red[wv] = ssq;
  __syncthreads();
  float tot = 0.f;
#pragma unroll
  for (int i = 0; i < 8; ++i) tot += red[i];
  float rstd = rsqrtf(tot / (float)D_ + 1e-6f);
  for (int d = tid; d < D_; d += 256)
    out[(size_t)t * D_ + d] = (bf16_t)(row[d] * rstd * w[d]);
}

__global__ __launch_bounds__(256) void k_ln2(const float* __restrict__ x,
                                             const float* __restrict__ w,
                                             float* __restrict__ out32,
                                             bf16_t* __restrict__ outb) {
  int t = blockIdx.x, tid = threadIdx.x, lane = tid & 31, wv = tid >> 5;
  const float* row = x + (size_t)t * D_;
  float ssq = 0.f;
  for (int d = tid; d < D_; d += 256) { float v = row[d]; ssq += v * v; }
#pragma unroll
  for (int m = 16; m; m >>= 1) ssq += __shfl_xor(ssq, m, 32);
  __shared__ float red[8];
  if (lane == 0) red[wv] = ssq;
  __syncthreads();
  float tot = 0.f;
#pragma unroll
  for (int i = 0; i < 8; ++i) tot += red[i];
  float rstd = rsqrtf(tot / (float)D_ + 1e-6f);
  for (int d = tid; d < D_; d += 256) {
    float v = row[d] * rstd * w[d];
    out32[(size_t)t * D_ + d] = v;
    outb[(size_t)t * D_ + d] = (bf16_t)v;
  }
}

// q/k head RMSNorm + RoPE; grid = T * nh blocks of 128
__global__ __launch_bounds__(128) void k_qknorm_rope(const float* __restrict__ lin,
                                                     const float* __restrict__ nw,
                                                     const float* __restrict__ cosb,
                                                     const float* __restrict__ sinb,
                                                     bf16_t* __restrict__ dst, int nh) {
  int rowi = blockIdx.x;
  int tok = rowi / nh;
  int d = threadIdx.x, lane = d & 31, wv = d >> 5;
  float v = lin[(size_t)rowi * HD_ + d];
  float s = v * v;
#pragma unroll
  for (int m = 16; m; m >>= 1) s += __shfl_xor(s, m, 32);
  __shared__ float red[4];
  __shared__ float nb[128];
  if (lane == 0) red[wv] = s;
  __syncthreads();
  float tot = red[0] + red[1] + red[2] + red[3];
  float n = v * rsqrtf(tot / (float)HD_ + 1e-6f) * nw[d];
  nb[d] = n;
  __syncthreads();
  float rot = (d < 64) ? -nb[d + 64] : nb[d - 64];
  float c  = cosb[(size_t)tok * HD_ + d];
  float sn = sinb[(size_t)tok * HD_ + d];
  dst[(size_t)rowi * HD_ + d] = (bf16_t)(n * c + rot * sn);
}

__global__ __launch_bounds__(256) void k_cvt_bf16(const float* __restrict__ in,
                                                  bf16_t* __restrict__ out, int n) {
  int i = blockIdx.x * 256 + threadIdx.x;
  if (i < n) out[i] = (bf16_t)in[i];
}

__global__ void k_zero_counts(int* c) { c[threadIdx.x] = 0; }

// ---------------- generic bf16 WMMA GEMM: C[MxN] = A(bf16,row) x Bw(f32,KxN) (+resid) ----------------
// Block tile 64x128, 8 waves as 4x2 groups, each wave: 1 A-frag x 4 B-frags -> 4 WMMA / k-step.
// A tile staged by TDM (async DMA, LDS row padding 64B->+16B); B staged+converted by all threads.
__global__ __launch_bounds__(256) void k_gemm_bf16(const bf16_t* __restrict__ A,
                                                   const float* __restrict__ Bw,
                                                   const float* __restrict__ resid,
                                                   float* __restrict__ C,
                                                   int N, int Kd) {
  __shared__ __align__(16) bf16_t As[64 * LDA];
  __shared__ __align__(16) bf16_t Bs[128 * LDA];
  int tid = threadIdx.x, lane = tid & 31, wv = tid >> 5;
  int m0 = blockIdx.x * 64, n0 = blockIdx.y * 128;
  int tr = wv & 3, tcg = (wv >> 2) * 4;
  const u32 asAddr = lds_addr_of(As);
  v8f acc[4];
#pragma unroll
  for (int j = 0; j < 4; ++j) acc[j] = zero8();
  for (int kb = 0; kb < Kd; kb += 32) {
    __syncthreads();
    if (wv == 0)
      tdm_load_bf16_2d(asAddr, A + (size_t)m0 * Kd + kb, 32, 64, 64, (u32)Kd, 3, 3);
#pragma unroll
    for (int it = 0; it < 4; ++it) { // stage B transposed: 32k x 128n f32 -> bf16 [n][k]
      int idx = tid + it * 256;      // float4 index, 1024 total
      int k = idx >> 5, nq = idx & 31;
      const float* src = Bw + (size_t)(kb + k) * N + n0 + nq * 4;
      float4 f = *(const float4*)src;
      if (kb + 32 < Kd) __builtin_prefetch(src + (size_t)32 * N, 0, 1);
      Bs[(nq * 4 + 0) * LDA + k] = (bf16_t)f.x;
      Bs[(nq * 4 + 1) * LDA + k] = (bf16_t)f.y;
      Bs[(nq * 4 + 2) * LDA + k] = (bf16_t)f.z;
      Bs[(nq * 4 + 3) * LDA + k] = (bf16_t)f.w;
    }
    if (wv == 0) __builtin_amdgcn_s_wait_tensorcnt(0);
    __syncthreads();
    v16bf a = load_a(As + tr * 16 * LDA, LDA, lane);
#pragma unroll
    for (int j = 0; j < 4; ++j) {
      v16bf b = load_bt(Bs + (tcg + j) * 16 * LDA, LDA, lane);
      acc[j] = wmma_bf16(a, b, acc[j]);
    }
  }
  int hi = lane >> 4, ln = lane & 15;
#pragma unroll
  for (int j = 0; j < 4; ++j) {
#pragma unroll
    for (int i = 0; i < 8; ++i) {
      int m = m0 + tr * 16 + i + 8 * hi;
      int n = n0 + (tcg + j) * 16 + ln;
      float v0 = acc[j][i];
      if (resid) v0 += resid[(size_t)m * N + n];
      C[(size_t)m * N + n] = v0;
    }
  }
}

// ---------------- flash attention (causal, GQA) ----------------
__global__ __launch_bounds__(128) void k_attn(const bf16_t* __restrict__ Q,
                                              const bf16_t* __restrict__ Kc,
                                              const bf16_t* __restrict__ Vc,
                                              bf16_t* __restrict__ Ob) {
  int bh = blockIdx.x;
  int b = bh / H_, h = bh % H_;
  int kvh = h / (H_ / KVH_);
  int qt = blockIdx.y;
  int tid = threadIdx.x, lane = tid & 31, wv = tid >> 5;
  int hi = lane >> 4, ln = lane & 15;

  __shared__ __align__(16) bf16_t Qs[64 * LQ];
  __shared__ __align__(16) bf16_t Ks[32 * LQ];
  __shared__ __align__(16) bf16_t Vt[128 * LDA];
  __shared__ __align__(16) bf16_t Ps[4][16 * LDA];

  // stage Q tile 64x128 via TDM (row stride H*HD, LDS pad 256B->+16B)
  if (wv == 0) {
    tdm_load_bf16_2d(lds_addr_of(Qs),
                     Q + ((size_t)(b * S_ + qt * 64) * H_ + h) * HD_,
                     128, 64, 64, (u32)(H_ * HD_), 5, 3);
    __builtin_amdgcn_s_wait_tensorcnt(0);
  }
  __syncthreads();
  v16bf qf[4];
#pragma unroll
  for (int ks = 0; ks < 4; ++ks) qf[ks] = load_a(Qs + wv * 16 * LQ + ks * 32, LQ, lane);

  v8f oacc[8];
#pragma unroll
  for (int t2 = 0; t2 < 8; ++t2) oacc[t2] = zero8();
  float mrun[8], lrun[8];
#pragma unroll
  for (int i = 0; i < 8; ++i) { mrun[i] = -1e30f; lrun[i] = 0.f; }

  int q0w = qt * 64 + wv * 16;
  int ntiles = qt * 2 + 2;
  const float scale = 0.08838834764831845f; // 1/sqrt(128)
  const u32 ksAddr = lds_addr_of(Ks);

  for (int kt = 0; kt < ntiles; ++kt) {
    int k0 = kt * 32;
    __syncthreads();
    if (wv == 0) // stage K (row-major == B^T for scores) via TDM
      tdm_load_bf16_2d(ksAddr,
                       Kc + ((size_t)(b * S_ + k0) * KVH_ + kvh) * HD_,
                       128, 32, 32, (u32)(KVH_ * HD_), 5, 3);
    { // stage V transposed: Vt[hd][key]
      int key = tid >> 2, part = tid & 3;
      const bf16_t* src = Vc + ((size_t)(b * S_ + k0 + key) * KVH_ + kvh) * HD_ + part * 32;
#pragma unroll
      for (int j = 0; j < 32; ++j) Vt[(part * 32 + j) * LDA + key] = src[j];
    }
    if (wv == 0) __builtin_amdgcn_s_wait_tensorcnt(0);
    __syncthreads();
    v8f c0 = zero8(), c1 = zero8();
#pragma unroll
    for (int ks = 0; ks < 4; ++ks) {
      v16bf bk0 = load_bt(Ks + ks * 32, LQ, lane);
      v16bf bk1 = load_bt(Ks + 16 * LQ + ks * 32, LQ, lane);
      c0 = wmma_bf16(qf[ks], bk0, c0);
      c1 = wmma_bf16(qf[ks], bk1, c1);
    }
#pragma unroll
    for (int i = 0; i < 8; ++i) {
      int m = i + 8 * hi;
      int q = q0w + m;
      float s0 = c0[i] * scale; if (k0 + ln > q)      s0 = -1e30f;
      float s1 = c1[i] * scale; if (k0 + 16 + ln > q) s1 = -1e30f;
      float mt = rowmax16(fmaxf(s0, s1));
      float mn = fmaxf(mrun[i], mt);
      float corr = __expf(mrun[i] - mn);
      float p0 = __expf(s0 - mn), p1 = __expf(s1 - mn);
      float rs = rowsum16(p0 + p1);
      lrun[i] = lrun[i] * corr + rs;
      mrun[i] = mn;
#pragma unroll
      for (int t2 = 0; t2 < 8; ++t2) oacc[t2][i] *= corr;
      Ps[wv][m * LDA + ln]      = (bf16_t)p0;
      Ps[wv][m * LDA + 16 + ln] = (bf16_t)p1;
    }
    // same-wave LDS write->read is in-order
    v16bf pa = load_a(Ps[wv], LDA, lane);
#pragma unroll
    for (int dj = 0; dj < 8; ++dj) {
      v16bf vb = load_bt(Vt + dj * 16 * LDA, LDA, lane);
      oacc[dj] = wmma_bf16(pa, vb, oacc[dj]);
    }
  }
#pragma unroll
  for (int i = 0; i < 8; ++i) {
    float inv = 1.0f / lrun[i];
    int q = q0w + i + 8 * hi;
    size_t base = ((size_t)(b * S_ + q)) * (H_ * HD_) + (size_t)h * HD_;
#pragma unroll
    for (int dj = 0; dj < 8; ++dj)
      Ob[base + dj * 16 + ln] = (bf16_t)(oacc[dj][i] * inv);
  }
}

// ---------------- router: fp32 logits, softmax, top-8 ----------------
__global__ __launch_bounds__(64) void k_router(const float* __restrict__ ht,
                                               const float* __restrict__ rw,
                                               int* __restrict__ topi,
                                               float* __restrict__ topv,
                                               int* __restrict__ counts) {
  int t = blockIdx.x, tid = threadIdx.x;
  float acc = 0.f;
  const float* row = ht + (size_t)t * D_;
  for (int d = 0; d < D_; ++d) acc += row[d] * rw[(size_t)d * E_ + tid];
  __shared__ float sp[64];
  __shared__ int   sti[8];
  sp[tid] = acc;
  __syncthreads();
  float mx = -1e30f;
  for (int i = 0; i < E_; ++i) mx = fmaxf(mx, sp[i]);
  float p = __expf(acc - mx);
  __syncthreads();
  sp[tid] = p;
  __syncthreads();
  float sum = 0.f;
  for (int i = 0; i < E_; ++i) sum += sp[i];
  __syncthreads();
  sp[tid] = p / sum;
  __syncthreads();
  if (tid == 0) {
    float sum8 = 0.f, stv[8];
    for (int kk = 0; kk < KT_; ++kk) {
      int best = 0; float bv = -1.f;
      for (int i = 0; i < E_; ++i) if (sp[i] > bv) { bv = sp[i]; best = i; }
      sti[kk] = best; stv[kk] = bv; sp[best] = -2.f; sum8 += bv;
    }
    float inv = 1.f / sum8;
    for (int kk = 0; kk < KT_; ++kk) {
      topi[(size_t)t * KT_ + kk] = sti[kk];
      topv[(size_t)t * KT_ + kk] = stv[kk] * inv;
    }
  }
  __syncthreads();
  if (tid < KT_) atomicAdd(&counts[sti[tid]], 1);
}

__global__ void k_scan(const int* __restrict__ counts, int* __restrict__ offsets,
                       int* __restrict__ cursors) {
  int tid = threadIdx.x;
  cursors[tid] = 0;
  if (tid == 0) {
    int run = 0;
    for (int e = 0; e < E_; ++e) { offsets[e] = run; run += counts[e]; }
  }
}

__global__ __launch_bounds__(256) void k_scatter(const int* __restrict__ topi,
                                                 const float* __restrict__ topv,
                                                 const int* __restrict__ offsets,
                                                 int* __restrict__ cursors,
                                                 int* __restrict__ tok_list,
                                                 float* __restrict__ wt_list) {
  int i = blockIdx.x * 256 + threadIdx.x; // < T*K
  int t = i >> 3;
  int e = topi[i];
  int pos = atomicAdd(&cursors[e], 1);
  int g = offsets[e] + pos;
  tok_list[g] = t;
  wt_list[g] = topv[i];
}

// ---------------- grouped MoE GEMM 1: act = silu(x@Wg) * (x@Wu) ----------------
// 64x128 tile, gathered A rows (manual stage), dual weight tiles -> 8 WMMA / wave / k-step.
__global__ __launch_bounds__(256) void k_moe_gemm1(const bf16_t* __restrict__ ht,
                                                   const float* __restrict__ wg,
                                                   const float* __restrict__ wu,
                                                   const int* __restrict__ tok_list,
                                                   const int* __restrict__ counts,
                                                   const int* __restrict__ offsets,
                                                   bf16_t* __restrict__ act) {
  int e = blockIdx.x, n0 = blockIdx.y * 128;
  int cnt = counts[e], off = offsets[e];
  int tid = threadIdx.x, lane = tid & 31, wv = tid >> 5;
  int tr = wv & 3, tcg = (wv >> 2) * 4;
  int hi = lane >> 4, ln = lane & 15;
  __shared__ __align__(16) bf16_t As[64 * LDA];
  __shared__ __align__(16) bf16_t Bg[128 * LDA];
  __shared__ __align__(16) bf16_t Bu[128 * LDA];
  const float* wgb = wg + (size_t)e * D_ * F_;
  const float* wub = wu + (size_t)e * D_ * F_;
  for (int r0 = 0; r0 < cnt; r0 += 64) {
    v8f cg[4], cu[4];
#pragma unroll
    for (int j = 0; j < 4; ++j) { cg[j] = zero8(); cu[j] = zero8(); }
    for (int kb = 0; kb < D_; kb += 32) {
      __syncthreads();
      { // gathered A rows
        int r = tid >> 2, ko = (tid & 3) * 8;
        int lr = r0 + r;
        int tok = tok_list[(lr < cnt) ? (off + lr) : off];
        uint4 va = *(const uint4*)(ht + (size_t)tok * D_ + kb + ko);
        *(uint4*)(As + r * LDA + ko) = va;
      }
#pragma unroll
      for (int it = 0; it < 4; ++it) {
        int idx = tid + it * 256;
        int k = idx >> 5, nq = idx & 31;
        const float* sg = wgb + (size_t)(kb + k) * F_ + n0 + nq * 4;
        const float* su = wub + (size_t)(kb + k) * F_ + n0 + nq * 4;
        float4 fg = *(const float4*)sg;
        float4 fu = *(const float4*)su;
        if (kb + 32 < D_) {
          __builtin_prefetch(sg + (size_t)32 * F_, 0, 1);
          __builtin_prefetch(su + (size_t)32 * F_, 0, 1);
        }
        Bg[(nq * 4 + 0) * LDA + k] = (bf16_t)fg.x;
        Bg[(nq * 4 + 1) * LDA + k] = (bf16_t)fg.y;
        Bg[(nq * 4 + 2) * LDA + k] = (bf16_t)fg.z;
        Bg[(nq * 4 + 3) * LDA + k] = (bf16_t)fg.w;
        Bu[(nq * 4 + 0) * LDA + k] = (bf16_t)fu.x;
        Bu[(nq * 4 + 1) * LDA + k] = (bf16_t)fu.y;
        Bu[(nq * 4 + 2) * LDA + k] = (bf16_t)fu.z;
        Bu[(nq * 4 + 3) * LDA + k] = (bf16_t)fu.w;
      }
      __syncthreads();
      v16bf a = load_a(As + tr * 16 * LDA, LDA, lane);
#pragma unroll
      for (int j = 0; j < 4; ++j) {
        v16bf bg = load_bt(Bg + (tcg + j) * 16 * LDA, LDA, lane);
        cg[j] = wmma_bf16(a, bg, cg[j]);
        v16bf bu = load_bt(Bu + (tcg + j) * 16 * LDA, LDA, lane);
        cu[j] = wmma_bf16(a, bu, cu[j]);
      }
    }
#pragma unroll
    for (int j = 0; j < 4; ++j) {
#pragma unroll
      for (int i = 0; i < 8; ++i) {
        int m = tr * 16 + i + 8 * hi;
        if (r0 + m < cnt) {
          size_t row = (size_t)(off + r0 + m);
          float g0 = cg[j][i], u0 = cu[j][i];
          act[row * F_ + n0 + (tcg + j) * 16 + ln] =
              (bf16_t)(g0 / (1.f + __expf(-g0)) * u0);
        }
      }
    }
  }
}

// ---------------- grouped MoE GEMM 2: out += wt * (act @ Wd) ----------------
__global__ __launch_bounds__(256) void k_moe_gemm2(const bf16_t* __restrict__ act,
                                                   const float* __restrict__ wd,
                                                   const int* __restrict__ tok_list,
                                                   const float* __restrict__ wt_list,
                                                   const int* __restrict__ counts,
                                                   const int* __restrict__ offsets,
                                                   float* __restrict__ out) {
  int e = blockIdx.x, n0 = blockIdx.y * 128;
  int cnt = counts[e], off = offsets[e];
  int tid = threadIdx.x, lane = tid & 31, wv = tid >> 5;
  int tr = wv & 3, tcg = (wv >> 2) * 4;
  int hi = lane >> 4, ln = lane & 15;
  __shared__ __align__(16) bf16_t As[64 * LDA];
  __shared__ __align__(16) bf16_t Bs[128 * LDA];
  const float* wdb = wd + (size_t)e * F_ * D_;
  const u32 asAddr = lds_addr_of(As);
  for (int r0 = 0; r0 < cnt; r0 += 64) {
    u32 rows = (u32)((cnt - r0 < 64) ? (cnt - r0) : 64); // clamp: OOB rows read zero
    v8f acc[4];
#pragma unroll
    for (int j = 0; j < 4; ++j) acc[j] = zero8();
    for (int kb = 0; kb < F_; kb += 32) {
      __syncthreads();
      if (wv == 0) // A rows are contiguous per expert -> TDM
        tdm_load_bf16_2d(asAddr, act + (size_t)(off + r0) * F_ + kb,
                         32, 64, rows, (u32)F_, 3, 3);
#pragma unroll
      for (int it = 0; it < 4; ++it) {
        int idx = tid + it * 256;
        int k = idx >> 5, nq = idx & 31;
        const float* src = wdb + (size_t)(kb + k) * D_ + n0 + nq * 4;
        float4 f = *(const float4*)src;
        if (kb + 32 < F_) __builtin_prefetch(src + (size_t)32 * D_, 0, 1);
        Bs[(nq * 4 + 0) * LDA + k] = (bf16_t)f.x;
        Bs[(nq * 4 + 1) * LDA + k] = (bf16_t)f.y;
        Bs[(nq * 4 + 2) * LDA + k] = (bf16_t)f.z;
        Bs[(nq * 4 + 3) * LDA + k] = (bf16_t)f.w;
      }
      if (wv == 0) __builtin_amdgcn_s_wait_tensorcnt(0);
      __syncthreads();
      v16bf a = load_a(As + tr * 16 * LDA, LDA, lane);
#pragma unroll
      for (int j = 0; j < 4; ++j) {
        v16bf b = load_bt(Bs + (tcg + j) * 16 * LDA, LDA, lane);
        acc[j] = wmma_bf16(a, b, acc[j]);
      }
    }
#pragma unroll
    for (int j = 0; j < 4; ++j) {
#pragma unroll
      for (int i = 0; i < 8; ++i) {
        int m = tr * 16 + i + 8 * hi;
        if (r0 + m < cnt) {
          int g = off + r0 + m;
          int tok = tok_list[g];
          float wt = wt_list[g];
          atomicAdd(&out[(size_t)tok * D_ + n0 + (tcg + j) * 16 + ln], wt * acc[j][i]);
        }
      }
    }
  }
}

// ---------------- host ----------------
extern "C" void kernel_launch(void* const* d_in, const int* in_sizes, int n_in,
                              void* d_out, int out_size, void* d_ws, size_t ws_size,
                              hipStream_t stream) {
  (void)in_sizes; (void)n_in; (void)out_size; (void)ws_size;
  const float* x    = (const float*)d_in[0];
  const float* cosb = (const float*)d_in[1];
  const float* sinb = (const float*)d_in[2];
  const float* ln1w = (const float*)d_in[3];
  const float* wq   = (const float*)d_in[4];
  const float* wk   = (const float*)d_in[5];
  const float* wvp  = (const float*)d_in[6];
  const float* wo   = (const float*)d_in[7];
  const float* qnw  = (const float*)d_in[8];
  const float* knw  = (const float*)d_in[9];
  const float* ln2w = (const float*)d_in[10];
  const float* rw   = (const float*)d_in[11];
  const float* wg   = (const float*)d_in[12];
  const float* wu   = (const float*)d_in[13];
  const float* wd   = (const float*)d_in[14];
  float* out = (float*)d_out;

  char* p = (char*)d_ws;
  auto carve = [&](size_t bytes) -> void* {
    void* r = (void*)p;
    p += (bytes + 255) & ~(size_t)255;
    return r;
  };
  bf16_t* h_bf    = (bf16_t*)carve((size_t)T_ * D_ * 2);
  float*  q_lin   = (float*) carve((size_t)T_ * H_ * HD_ * 4);
  float*  k_lin   = (float*) carve((size_t)T_ * KVH_ * HD_ * 4);
  float*  v_lin   = (float*) carve((size_t)T_ * KVH_ * HD_ * 4);
  bf16_t* q_bf    = (bf16_t*)carve((size_t)T_ * H_ * HD_ * 2);
  bf16_t* k_bf    = (bf16_t*)carve((size_t)T_ * KVH_ * HD_ * 2);
  bf16_t* v_bf    = (bf16_t*)carve((size_t)T_ * KVH_ * HD_ * 2);
  bf16_t* o_bf    = (bf16_t*)carve((size_t)T_ * H_ * HD_ * 2);
  float*  ht32    = (float*) carve((size_t)T_ * D_ * 4);
  bf16_t* ht_bf   = (bf16_t*)carve((size_t)T_ * D_ * 2);
  int*    topi    = (int*)   carve((size_t)T_ * KT_ * 4);
  float*  topv    = (float*) carve((size_t)T_ * KT_ * 4);
  int*    counts  = (int*)   carve(E_ * 4);
  int*    offsets = (int*)   carve(E_ * 4);
  int*    cursors = (int*)   carve(E_ * 4);
  int*    tokl    = (int*)   carve((size_t)T_ * KT_ * 4);
  float*  wtl     = (float*) carve((size_t)T_ * KT_ * 4);
  bf16_t* act     = (bf16_t*)carve((size_t)T_ * KT_ * F_ * 2);

  k_zero_counts<<<1, E_, 0, stream>>>(counts);
  k_rmsnorm_bf16<<<T_, 256, 0, stream>>>(x, ln1w, h_bf);

  k_gemm_bf16<<<dim3(T_ / 64, (H_ * HD_) / 128), 256, 0, stream>>>(h_bf, wq, nullptr, q_lin, H_ * HD_, D_);
  k_gemm_bf16<<<dim3(T_ / 64, (KVH_ * HD_) / 128), 256, 0, stream>>>(h_bf, wk, nullptr, k_lin, KVH_ * HD_, D_);
  k_gemm_bf16<<<dim3(T_ / 64, (KVH_ * HD_) / 128), 256, 0, stream>>>(h_bf, wvp, nullptr, v_lin, KVH_ * HD_, D_);

  k_qknorm_rope<<<T_ * H_, 128, 0, stream>>>(q_lin, qnw, cosb, sinb, q_bf, H_);
  k_qknorm_rope<<<T_ * KVH_, 128, 0, stream>>>(k_lin, knw, cosb, sinb, k_bf, KVH_);
  k_cvt_bf16<<<(T_ * KVH_ * HD_) / 256, 256, 0, stream>>>(v_lin, v_bf, T_ * KVH_ * HD_);

  k_attn<<<dim3(BB * H_, S_ / 64), 128, 0, stream>>>(q_bf, k_bf, v_bf, o_bf);

  // out = x + o @ wo
  k_gemm_bf16<<<dim3(T_ / 64, D_ / 128), 256, 0, stream>>>(o_bf, wo, x, out, D_, H_ * HD_);

  k_ln2<<<T_, 256, 0, stream>>>(out, ln2w, ht32, ht_bf);
  k_router<<<T_, 64, 0, stream>>>(ht32, rw, topi, topv, counts);
  k_scan<<<1, E_, 0, stream>>>(counts, offsets, cursors);
  k_scatter<<<(T_ * KT_) / 256, 256, 0, stream>>>(topi, topv, offsets, cursors, tokl, wtl);

  k_moe_gemm1<<<dim3(E_, F_ / 128), 256, 0, stream>>>(ht_bf, wg, wu, tokl, counts, offsets, act);
  k_moe_gemm2<<<dim3(E_, D_ / 128), 256, 0, stream>>>(act, wd, tokl, wtl, counts, offsets, out);
}
